// MixupTrainLoss_72155450573201
// MI455X (gfx1250) — matrix operand order repaired and checked by reading, not compiled, not run on hardware
//
#include <hip/hip_runtime.h>
#include <hip/hip_bf16.h>

typedef _Float16 f16;
typedef _Float16 v8h  __attribute__((ext_vector_type(8)));
typedef _Float16 v16h __attribute__((ext_vector_type(16)));
typedef float    v8f  __attribute__((ext_vector_type(8)));
typedef __attribute__((address_space(3))) f16 lds_f16;

#define B_      512
#define D_      256
#define C_      100000
#define MARGIN  0.2f
#define SCALE   30.0f
#define LOG2E   1.44269504088896340736f
#define NCTILES (C_/16)      // 6250 column tiles
#define NCB     64           // column chunks (gridDim.y); partials per row
#define WPB     8            // waves per block (256 threads, wave32)
#define RPG     128          // rows per block (8 waves x 16-row tile)

// LDS B tile: 16 rows x 256 halves, padded to 264 halves (528B row stride).
// 528B = 132 dwords, 132 % 64 = 4 -> lane ln hits banks 4*ln.. : conflict-free.
#define LROW    264

// workspace byte offsets (all 256B aligned)
#define XN_OFF   ((size_t)0)
#define WN_OFF   (XN_OFF + (size_t)B_*D_*2)        // 262,144
#define PART_OFF (WN_OFF + (size_t)C_*D_*2)        // +51,200,000
#define SPEC_OFF (PART_OFF + (size_t)B_*NCB*4)     // +131,072

// ---------------------------------------------------------------------------
// Row-wise L2 normalize: one wave per row of 256 f32 -> f16
// ---------------------------------------------------------------------------
__global__ void __launch_bounds__(256)
normalize_rows_kernel(const float* __restrict__ in, f16* __restrict__ out, int rows) {
    int wid  = (blockIdx.x * blockDim.x + threadIdx.x) >> 5;
    int lane = threadIdx.x & 31;
    if (wid >= rows) return;
    const float* r = in + (size_t)wid * D_;
    float v[8];
    float ss = 0.f;
#pragma unroll
    for (int i = 0; i < 8; ++i) { v[i] = r[lane + i * 32]; ss += v[i] * v[i]; }
#pragma unroll
    for (int m = 1; m < 32; m <<= 1) ss += __shfl_xor(ss, m, 32);
    float inv = 1.0f / fmaxf(sqrtf(ss), 1e-12f);
    f16* o = out + (size_t)wid * D_;
#pragma unroll
    for (int i = 0; i < 8; ++i) o[lane + i * 32] = (f16)(v[i] * inv);
}

// ---------------------------------------------------------------------------
// Issue one B tile (16 rows x 512B) Global->LDS via async-to-LDS b128.
// 512 16B chunks, 256 threads -> 2 chunks per lane (ASYNCcnt += 2 per wave).
// ---------------------------------------------------------------------------
__device__ __forceinline__ void issue_btile_copy(const f16* __restrict__ wn,
                                                 int cbase, unsigned lds_base,
                                                 int t /*0..255*/) {
#pragma unroll
    for (int j = 0; j < 2; ++j) {
        int q    = t + j * 256;          // 0..511
        int row  = q >> 5;               // 0..15
        int off  = (q & 31);             // 16B chunk within the 512B row
        const f16* src = wn + (size_t)(cbase + row) * D_ + off * 8;
        unsigned   dst = lds_base + (unsigned)(row * (LROW * 2) + off * 16);
        asm volatile("global_load_async_to_lds_b128 %0, %1, off"
                     :: "v"(dst), "v"(src) : "memory");
    }
}

// ---------------------------------------------------------------------------
// Fused cos-GEMM + AM-softmax penalty + shifted exp-sum.
// Grid: (B/128 row groups, NCB col chunks), 256 threads (8 waves).
// Wave w owns the 16-row tile [rgbase + 16w, +16); its A fragments stay
// VGPR-resident for the whole sweep. All 8 waves share each 16-col B tile,
// double-buffered in LDS via global_load_async_to_lds_b128 / s_wait_asynccnt.
// ---------------------------------------------------------------------------
__global__ void __launch_bounds__(256)
gemm_softmax_kernel(const f16* __restrict__ xn, const f16* __restrict__ wn,
                    const int* __restrict__ t1, const int* __restrict__ p1,
                    const int* __restrict__ t2, const int* __restrict__ p2,
                    float* __restrict__ part, float* __restrict__ spec) {
    __shared__ f16 bbuf[2][16 * LROW];

    const int lane  = threadIdx.x & 31;
    const int wave  = threadIdx.x >> 5;
    const int g     = lane >> 4;      // half-wave group (C layout: M in [g*8, g*8+8))
    const int ln    = lane & 15;      // N index within tile / row index for frags
    const int rbase = blockIdx.x * RPG + wave * 16;

    const unsigned lds0 = (unsigned)(uintptr_t)(lds_f16*)&bbuf[0][0];
    const unsigned lds1 = (unsigned)(uintptr_t)(lds_f16*)&bbuf[1][0];

    // --- resident A fragments: 16-bit A 16x32 layout, lane group g holds
    //     K = kc*32 + g*8 + {0..7} and kc*32 + g*8 + 16 + {0..7}
    v16h afrag[8];
    {
        const f16* arow = xn + (size_t)(rbase + ln) * D_;
#pragma unroll
        for (int kc = 0; kc < 8; ++kc) {
            int k0 = kc * 32 + g * 8;
            union { v16h v; v8h h[2]; } u;
            u.h[0] = *(const v8h*)(arow + k0);
            u.h[1] = *(const v8h*)(arow + k0 + 16);
            afrag[kc] = u.v;
        }
    }

    // per-lane penalty indices for the 8 rows this lane's acc registers cover
    int t1v[8], p1v[8], t2v[8], p2v[8];
#pragma unroll
    for (int r = 0; r < 8; ++r) {
        int row = rbase + r + g * 8;
        t1v[r] = t1[row]; p1v[r] = p1[row]; t2v[r] = t2[row]; p2v[r] = p2[row];
    }

    float sm[8];
#pragma unroll
    for (int r = 0; r < 8; ++r) sm[r] = 0.f;

    const int t = threadIdx.x;
    const int y = blockIdx.y;

    // prologue: stage tiles y and y+NCB (double buffer)
    issue_btile_copy(wn, y * 16, lds0, t);
    if (y + NCB < NCTILES) issue_btile_copy(wn, (y + NCB) * 16, lds1, t);

    int cur = 0;
    for (int ct = y; ct < NCTILES; ct += NCB) {
        // retire this buffer's async copies (per wave), then block-sync
        if (ct + NCB < NCTILES) {
            asm volatile("s_wait_asynccnt 0x2" ::: "memory");
        } else {
            asm volatile("s_wait_asynccnt 0x0" ::: "memory");
        }
        __syncthreads();

        const f16* lrow = &bbuf[cur][ln * LROW];
        v8f acc = {};
#pragma unroll
        for (int kc = 0; kc < 8; ++kc) {
            // B 32x16 f16 layout: lane group g holds K = kc*32 + g*8 + {0..7} and kc*32 + g*8 + 16 + {0..7}
            int kb = kc * 32 + g * 8;
            union { v16h v; v8h h[2]; } u;
            u.h[0] = *(const v8h*)(lrow + kb);
            u.h[1] = *(const v8h*)(lrow + kb + 16);
            acc = __builtin_amdgcn_wmma_f32_16x16x32_f16(
                false, afrag[kc], false, u.v, (short)0, acc, false, false);
        }

        // post-process: acc[r] = cos for (row = rbase + r + g*8, col = cbase + ln)
        const int col = ct * 16 + ln;
#pragma unroll
        for (int r = 0; r < 8; ++r) {
            float c   = acc[r];
            float val = SCALE * c;
            bool i1 = (col == t1v[r]);
            bool i2 = (col == p1v[r]);
            bool i3 = (col == t2v[r]);
            bool i4 = (col == p2v[r]);
            // reference overwrite order: t1 (scaled) then p1, t2, p2 (unscaled)
            if (i1) val = SCALE * (c - MARGIN);
            if (i2) val = c - MARGIN;
            if (i3) val = c - MARGIN;
            if (i4) val = c - MARGIN;
            if (i1 | i2 | i3 | i4) {
                int row = rbase + r + g * 8;
                if (i1) spec[row * 4 + 0] = val;
                if (i2) spec[row * 4 + 1] = val;
                if (i3) spec[row * 4 + 2] = val;
                if (i4) spec[row * 4 + 3] = val;
            }
            // fixed-shift exp: val in [-36, 30] so exp(val-30) is safe in f32
            sm[r] += exp2f(fmaf(val, LOG2E, -SCALE * LOG2E));
        }

        // everyone done reading buf[cur]; refill it with tile ct + 2*NCB
        __syncthreads();
        if (ct + 2 * NCB < NCTILES)
            issue_btile_copy(wn, (ct + 2 * NCB) * 16, cur ? lds1 : lds0, t);
        cur ^= 1;
    }

    // reduce exp-sums across the 16 lanes of each half-wave (same M rows)
#pragma unroll
    for (int r = 0; r < 8; ++r) {
#pragma unroll
        for (int m = 1; m < 16; m <<= 1) sm[r] += __shfl_xor(sm[r], m, 32);
    }
    if (ln == 0) {
#pragma unroll
        for (int r = 0; r < 8; ++r) {
            int row = rbase + r + g * 8;
            part[(size_t)row * NCB + y] = sm[r];
        }
    }
}

// ---------------------------------------------------------------------------
// Finalize: per-row logsumexp merge + weighted cross-entropy -> scalar loss
// ---------------------------------------------------------------------------
__global__ void __launch_bounds__(512)
finalize_kernel(const float* __restrict__ part, const float* __restrict__ spec,
                const float* __restrict__ lamp, float* __restrict__ out) {
    __shared__ float red[512];
    int b = threadIdx.x;
    float s = 0.f;
    for (int j = 0; j < NCB; ++j) s += part[(size_t)b * NCB + j];
    float lse = logf(s) + SCALE;               // undo the fixed shift
    float l0 = spec[b * 4 + 0] - lse;
    float l1 = spec[b * 4 + 1] - lse;
    float l2 = spec[b * 4 + 2] - lse;
    float l3 = spec[b * 4 + 3] - lse;
    float lam = lamp[0];
    float contrib = lam * (0.2f * (-l0) + 0.8f * (-l1))
                  + (1.f - lam) * (0.2f * (-l2) + 0.8f * (-l3));
    red[b] = contrib;
    __syncthreads();
#pragma unroll
    for (int st = 256; st > 0; st >>= 1) {
        if (b < st) red[b] += red[b + st];
        __syncthreads();
    }
    if (b == 0) out[0] = red[0] / (float)B_;
}

// ---------------------------------------------------------------------------
extern "C" void kernel_launch(void* const* d_in, const int* in_sizes, int n_in,
                              void* d_out, int out_size, void* d_ws, size_t ws_size,
                              hipStream_t stream) {
    (void)in_sizes; (void)n_in; (void)out_size; (void)ws_size;
    const float* x   = (const float*)d_in[0];   // [512,256,1] f32
    const float* w   = (const float*)d_in[1];   // [100000,256] f32
    const float* lam = (const float*)d_in[2];   // scalar f32
    const int*   t1  = (const int*)d_in[3];
    const int*   p1  = (const int*)d_in[4];
    const int*   t2  = (const int*)d_in[5];
    const int*   p2  = (const int*)d_in[6];

    char* ws = (char*)d_ws;
    f16*   xn   = (f16*)(ws + XN_OFF);
    f16*   wn   = (f16*)(ws + WN_OFF);
    float* part = (float*)(ws + PART_OFF);
    float* spec = (float*)(ws + SPEC_OFF);

    normalize_rows_kernel<<<B_ / 8, 256, 0, stream>>>(x, xn, B_);
    normalize_rows_kernel<<<C_ / 8, 256, 0, stream>>>(w, wn, C_);

    dim3 grid(B_ / RPG, NCB);                   // (4, 64)
    gemm_softmax_kernel<<<grid, 256, 0, stream>>>(xn, wn, t1, p1, t2, p2, part, spec);

    finalize_kernel<<<1, 512, 0, stream>>>(part, spec, lam, (float*)d_out);
}
